// AutoregressivePointerNet_45122926412241
// MI455X (gfx1250) — matrix operand order repaired and checked by reading, not compiled, not run on hardware
//
#include <hip/hip_runtime.h>
#include <hip/hip_fp16.h>

#define B_   256
#define S_   2048
#define H_   128
#define G4   512     // 4*H
#define T_   64      // target_len
#define WP   136     // padded LDS row length in halves (17*8 -> conflict-free)
#define XC   128     // x prefetch chunk (steps)

typedef __attribute__((ext_vector_type(16))) _Float16 v16h;
typedef __attribute__((ext_vector_type(8)))  _Float16 v8h;
typedef __attribute__((ext_vector_type(8)))  float    v8f;

__device__ __forceinline__ float fast_tanh(float x) {
#if __has_builtin(__builtin_amdgcn_tanhf)
  return __builtin_amdgcn_tanhf(x);
#else
  return tanhf(x);
#endif
}
__device__ __forceinline__ float sigm(float x) { return 1.0f / (1.0f + __expf(-x)); }

// A fragment (16x32 f16): lane L: M=L&15, halves [0..7]=A[M][k0..k0+7],
// [8..15]=A[M][k0+16..k0+23], k0 = kt*32 + (L>=16 ? 8 : 0). p = &A[M][k0].
__device__ __forceinline__ v16h ldsA(const _Float16* p) {
  v8h lo = *(const v8h*)(p);
  v8h hi = *(const v8h*)(p + 16);
  v16h r;
#pragma unroll
  for (int e = 0; e < 8; ++e) { r[e] = lo[e]; r[8 + e] = hi[e]; }
  return r;
}
// B fragment (32x16 f16): lane L: n=L&15, halves [0..7]=B[k0..k0+7][n],
// [8..15]=B[k0+16..k0+23][n], k0 = kt*32 + (L>=16 ? 8 : 0). Stored as W[j][k]
// row-major, j = tile*16+n; p = &W[j][kt*32 + (L>=16 ? 16 : 0)].
__device__ __forceinline__ v16h ldsB(const _Float16* p) {
  p -= ((threadIdx.x & 31) >> 4) * 8;
  v8h lo = *(const v8h*)(p);
  v8h hi = *(const v8h*)(p + 16);
  v16h r;
#pragma unroll
  for (int e = 0; e < 8; ++e) { r[e] = lo[e]; r[8 + e] = hi[e]; }
  return r;
}
__device__ __forceinline__ v8f wmma16(v16h a, v16h b, v8f c) {
  return __builtin_amdgcn_wmma_f32_16x16x32_f16(false, a, false, b, (short)0, c,
                                                false, false);
}

// ---------------------------------------------------------------- encoder ---
// grid 16 x 256 threads. Block owns batch rows [b0, b0+16). 2048 steps inside.
__global__ __launch_bounds__(256)
void enc_kernel(const float* __restrict__ x, const float* __restrict__ Wih,
                const float* __restrict__ Whh, const float* __restrict__ bih,
                const float* __restrict__ bhh, const float* __restrict__ W1w,
                const float* __restrict__ W1b, _Float16* __restrict__ keys,
                float* __restrict__ hx, float* __restrict__ cx) {
  extern __shared__ char smem[];
  _Float16* whh = (_Float16*)smem;           // [512][WP] f16
  _Float16* w1  = whh + G4 * WP;             // [128][WP] f16
  _Float16* hb  = w1 + H_ * WP;              // [2][16][WP] f16 (double buffer)
  float*    xc  = (float*)(hb + 2 * 16 * WP); // [XC][16] f32

  const int tid = threadIdx.x;
  const int wv = tid >> 5, lane = tid & 31;
  const int n = lane & 15, hlf = lane >> 4;
  const int b0 = blockIdx.x * 16;

  for (int i = tid; i < G4 * H_; i += 256) {
    int j = i >> 7, k = i & 127;
    whh[j * WP + k] = (_Float16)Whh[i];
  }
  for (int i = tid; i < H_ * H_; i += 256) {
    int j = i >> 7, k = i & 127;
    w1[j * WP + k] = (_Float16)W1w[i];
  }
  for (int i = tid; i < 2 * 16 * WP; i += 256) hb[i] = (_Float16)0.f;

  const int jh = wv * 16 + n;  // hidden column this lane owns
  const float bi = bih[jh] + bhh[jh];
  const float bf = bih[H_ + jh] + bhh[H_ + jh];
  const float bg = bih[2 * H_ + jh] + bhh[2 * H_ + jh];
  const float bo = bih[3 * H_ + jh] + bhh[3 * H_ + jh];
  const float wxi = Wih[jh], wxf = Wih[H_ + jh];
  const float wxg = Wih[2 * H_ + jh], wxo = Wih[3 * H_ + jh];
  const float bk1 = W1b[jh];

  v8f c = {};
  int p = 0;
  __syncthreads();

  for (int t = 0; t < S_; ++t) {
    if ((t & (XC - 1)) == 0) {
      __syncthreads();
      for (int i = tid; i < XC * 16; i += 256) {
        int tt = i & (XC - 1), bb = i >> 7;
        xc[tt * 16 + bb] = x[(size_t)(b0 + bb) * S_ + t + tt];
      }
    }
    __syncthreads();

    const _Float16* hB = hb + p * 16 * WP;  // h(t)
    v16h a[4];
#pragma unroll
    for (int kt = 0; kt < 4; ++kt)
      a[kt] = ldsA(hB + n * WP + kt * 32 + hlf * 8);

    v8f aI = {}, aF = {}, aG = {}, aO = {}, aK = {};
#pragma unroll
    for (int kt = 0; kt < 4; ++kt) {
      const int ko = kt * 32 + hlf * 16;
      v16h vb;
      vb = ldsB(whh + (0 * H_ + jh) * WP + ko); aI = wmma16(a[kt], vb, aI);
      vb = ldsB(whh + (1 * H_ + jh) * WP + ko); aF = wmma16(a[kt], vb, aF);
      vb = ldsB(whh + (2 * H_ + jh) * WP + ko); aG = wmma16(a[kt], vb, aG);
      vb = ldsB(whh + (3 * H_ + jh) * WP + ko); aO = wmma16(a[kt], vb, aO);
      vb = ldsB(w1 + jh * WP + ko);             aK = wmma16(a[kt], vb, aK);
    }
    // aK = W1*h(t) = keys column (t-1)  (enc_out[t-1] == h(t))
    if (t > 0) {
#pragma unroll
      for (int r = 0; r < 8; ++r) {
        int M = r + hlf * 8;
        keys[((size_t)(b0 + M) * S_ + (t - 1)) * H_ + jh] =
            (_Float16)(aK[r] + bk1);
      }
    }
    _Float16* hN = hb + (p ^ 1) * 16 * WP;
#pragma unroll
    for (int r = 0; r < 8; ++r) {
      int M = r + hlf * 8;
      float xv = xc[(t & (XC - 1)) * 16 + M];
      float iv = sigm(aI[r] + bi + xv * wxi);
      float fv = sigm(aF[r] + bf + xv * wxf);
      float gv = fast_tanh(aG[r] + bg + xv * wxg);
      float ov = sigm(aO[r] + bo + xv * wxo);
      float cv = fv * c[r] + iv * gv;
      c[r] = cv;
      float hv = ov * fast_tanh(cv);
      hN[M * WP + jh] = (_Float16)hv;
      if (t == S_ - 1) {
        hx[(size_t)(b0 + M) * H_ + jh] = hv;
        cx[(size_t)(b0 + M) * H_ + jh] = cv;
      }
    }
    p ^= 1;
  }
  // tail: keys column S-1 from h(S)
  __syncthreads();
  {
    const _Float16* hB = hb + p * 16 * WP;
    v8f aK = {};
#pragma unroll
    for (int kt = 0; kt < 4; ++kt) {
      v16h a = ldsA(hB + n * WP + kt * 32 + hlf * 8);
      v16h vb = ldsB(w1 + jh * WP + kt * 32 + hlf * 16);
      aK = wmma16(a, vb, aK);
    }
#pragma unroll
    for (int r = 0; r < 8; ++r) {
      int M = r + hlf * 8;
      keys[((size_t)(b0 + M) * S_ + (S_ - 1)) * H_ + jh] =
          (_Float16)(aK[r] + bk1);
    }
  }
}

// ----------------------------------------------------- decoder LSTM + query --
__global__ __launch_bounds__(256)
void dec_step_kernel(const float* __restrict__ dWih, const float* __restrict__ dWhh,
                     const float* __restrict__ dbih, const float* __restrict__ dbhh,
                     const float* __restrict__ W2w, const float* __restrict__ W2b,
                     const float* __restrict__ dec_x, float* __restrict__ hx,
                     float* __restrict__ cx, float* __restrict__ query) {
  extern __shared__ char smem[];
  _Float16* whh = (_Float16*)smem;    // [512][WP]
  _Float16* w2  = whh + G4 * WP;      // [128][WP]
  _Float16* hA  = w2 + H_ * WP;       // [16][WP]

  const int tid = threadIdx.x;
  const int wv = tid >> 5, lane = tid & 31;
  const int n = lane & 15, hlf = lane >> 4;
  const int b0 = blockIdx.x * 16;

  for (int i = tid; i < G4 * H_; i += 256) {
    int j = i >> 7, k = i & 127;
    whh[j * WP + k] = (_Float16)dWhh[i];
  }
  for (int i = tid; i < H_ * H_; i += 256) {
    int j = i >> 7, k = i & 127;
    w2[j * WP + k] = (_Float16)W2w[i];
  }
  for (int i = tid; i < 16 * H_; i += 256) {
    int bb = i >> 7, k = i & 127;
    hA[bb * WP + k] = (_Float16)hx[(size_t)(b0 + bb) * H_ + k];
  }
  const int jh = wv * 16 + n;
  const float bi = dbih[jh] + dbhh[jh];
  const float bf = dbih[H_ + jh] + dbhh[H_ + jh];
  const float bg = dbih[2 * H_ + jh] + dbhh[2 * H_ + jh];
  const float bo = dbih[3 * H_ + jh] + dbhh[3 * H_ + jh];
  const float wxi = dWih[jh], wxf = dWih[H_ + jh];
  const float wxg = dWih[2 * H_ + jh], wxo = dWih[3 * H_ + jh];
  const float b2 = W2b[jh];
  __syncthreads();

  v16h a[4];
#pragma unroll
  for (int kt = 0; kt < 4; ++kt)
    a[kt] = ldsA(hA + n * WP + kt * 32 + hlf * 8);

  v8f aI = {}, aF = {}, aG = {}, aO = {};
#pragma unroll
  for (int kt = 0; kt < 4; ++kt) {
    const int ko = kt * 32 + hlf * 16;
    v16h vb;
    vb = ldsB(whh + (0 * H_ + jh) * WP + ko); aI = wmma16(a[kt], vb, aI);
    vb = ldsB(whh + (1 * H_ + jh) * WP + ko); aF = wmma16(a[kt], vb, aF);
    vb = ldsB(whh + (2 * H_ + jh) * WP + ko); aG = wmma16(a[kt], vb, aG);
    vb = ldsB(whh + (3 * H_ + jh) * WP + ko); aO = wmma16(a[kt], vb, aO);
  }
  float hnew[8];
#pragma unroll
  for (int r = 0; r < 8; ++r) {
    int M = r + hlf * 8;
    float xv = dec_x[b0 + M];
    float c0 = cx[(size_t)(b0 + M) * H_ + jh];
    float iv = sigm(aI[r] + bi + xv * wxi);
    float fv = sigm(aF[r] + bf + xv * wxf);
    float gv = fast_tanh(aG[r] + bg + xv * wxg);
    float ov = sigm(aO[r] + bo + xv * wxo);
    float cv = fv * c0 + iv * gv;
    float hv = ov * fast_tanh(cv);
    cx[(size_t)(b0 + M) * H_ + jh] = cv;
    hx[(size_t)(b0 + M) * H_ + jh] = hv;
    hnew[r] = hv;
  }
  __syncthreads();
#pragma unroll
  for (int r = 0; r < 8; ++r) {
    int M = r + hlf * 8;
    hA[M * WP + jh] = (_Float16)hnew[r];
  }
  __syncthreads();
  v8f aQ = {};
#pragma unroll
  for (int kt = 0; kt < 4; ++kt) {
    v16h aa = ldsA(hA + n * WP + kt * 32 + hlf * 8);
    v16h vb = ldsB(w2 + jh * WP + kt * 32 + hlf * 16);
    aQ = wmma16(aa, vb, aQ);
  }
#pragma unroll
  for (int r = 0; r < 8; ++r) {
    int M = r + hlf * 8;
    query[(size_t)(b0 + M) * H_ + jh] = aQ[r] + b2;
  }
}

// ------------------------------------------------- attention + pointer pick --
// grid = B_ blocks (one batch row each), 256 threads.
__global__ __launch_bounds__(256)
void attn_kernel(const float* __restrict__ x, const _Float16* __restrict__ keys,
                 const float* __restrict__ query, const float* __restrict__ vtw,
                 const float* __restrict__ vtb, float* __restrict__ mask,
                 float* __restrict__ dec_x, float* __restrict__ out, int step) {
  extern __shared__ char smem[];
  _Float16* kch = (_Float16*)smem;            // [256][WP]
  float* q  = (float*)(kch + 256 * WP);       // [128]
  float* vt = q + H_;                          // [128]
  float* rv = vt + H_;                         // [256]
  int*   ri = (int*)(rv + 256);                // [256]

  const int tid = threadIdx.x;
  const int b = blockIdx.x;
  if (tid < H_) {
    q[tid] = query[(size_t)b * H_ + tid];
    vt[tid] = vtw[tid];
  }
  const float vb0 = vtb[0];
  __syncthreads();

  float sc[8];
#pragma unroll
  for (int ch = 0; ch < 8; ++ch) {
    __syncthreads();
    const _Float16* src = keys + ((size_t)b * S_ + ch * 256) * H_;
    for (int u = tid; u < 256 * H_ / 8; u += 256) {
      int row = u >> 4;          // 16 v8h per row
      int col = (u & 15) * 8;
      *(v8h*)(kch + row * WP + col) = *(const v8h*)(src + (size_t)row * H_ + col);
    }
    __syncthreads();
    int s = ch * 256 + tid;
    float acc = 0.f;
    const _Float16* kr = kch + tid * WP;
#pragma unroll
    for (int i = 0; i < 16; ++i) {
      v8h kk = *(const v8h*)(kr + i * 8);
#pragma unroll
      for (int e = 0; e < 8; ++e) {
        int j = i * 8 + e;
        acc += vt[j] * fast_tanh((float)kk[e] + q[j]);
      }
    }
    float mv = mask[(size_t)b * S_ + s];
    sc[ch] = (mv > 0.f) ? -1e9f : (acc + vb0);
  }
  // argmax (first-occurrence ties)
  float best = sc[0];
  int bi = tid;
#pragma unroll
  for (int ch = 1; ch < 8; ++ch) {
    int s = ch * 256 + tid;
    if (sc[ch] > best) { best = sc[ch]; bi = s; }
  }
  __syncthreads();
  rv[tid] = best; ri[tid] = bi;
  __syncthreads();
  for (int st = 128; st > 0; st >>= 1) {
    if (tid < st) {
      float ov = rv[tid + st]; int oi = ri[tid + st];
      if (ov > rv[tid] || (ov == rv[tid] && oi < ri[tid])) {
        rv[tid] = ov; ri[tid] = oi;
      }
    }
    __syncthreads();
  }
  const float m = rv[0];
  const int idx = ri[0];
  __syncthreads();
  float se = 0.f;
#pragma unroll
  for (int ch = 0; ch < 8; ++ch) se += __expf(sc[ch] - m);
  rv[tid] = se;
  __syncthreads();
  for (int st = 128; st > 0; st >>= 1) {
    if (tid < st) rv[tid] += rv[tid + st];
    __syncthreads();
  }
  if (tid == 0) {
    float lp = -__logf(rv[0]);  // score[idx]==m  ->  lp = -log(sum exp(s-m))
    out[(size_t)b * T_ + step] = (float)idx;
    out[(size_t)B_ * T_ + (size_t)b * T_ + step] = lp;
    mask[(size_t)b * S_ + idx] = 1.0f;
    dec_x[b] = x[(size_t)b * S_ + idx];
  }
}

__global__ __launch_bounds__(256)
void init_kernel(float* __restrict__ mask, float* __restrict__ dec_x) {
  size_t i = (size_t)blockIdx.x * blockDim.x + threadIdx.x;
  if (i < (size_t)B_ * S_) mask[i] = 0.f;
  if (i < B_) dec_x[i] = 0.f;
}

extern "C" void kernel_launch(void* const* d_in, const int* in_sizes, int n_in,
                              void* d_out, int out_size, void* d_ws, size_t ws_size,
                              hipStream_t stream) {
  const float* x    = (const float*)d_in[0];
  const float* eWih = (const float*)d_in[1];
  const float* eWhh = (const float*)d_in[2];
  const float* ebih = (const float*)d_in[3];
  const float* ebhh = (const float*)d_in[4];
  const float* dWih = (const float*)d_in[5];
  const float* dWhh = (const float*)d_in[6];
  const float* dbih = (const float*)d_in[7];
  const float* dbhh = (const float*)d_in[8];
  const float* W1w  = (const float*)d_in[9];
  const float* W1b  = (const float*)d_in[10];
  const float* W2w  = (const float*)d_in[11];
  const float* W2b  = (const float*)d_in[12];
  const float* vtw  = (const float*)d_in[13];
  const float* vtb  = (const float*)d_in[14];
  float* out = (float*)d_out;

  char* ws = (char*)d_ws;
  _Float16* keys = (_Float16*)ws;  ws += (size_t)B_ * S_ * H_ * 2;
  float* hx    = (float*)ws;       ws += (size_t)B_ * H_ * 4;
  float* cx    = (float*)ws;       ws += (size_t)B_ * H_ * 4;
  float* query = (float*)ws;       ws += (size_t)B_ * H_ * 4;
  float* mask  = (float*)ws;       ws += (size_t)B_ * S_ * 4;
  float* dec_x = (float*)ws;       ws += (size_t)B_ * 4;

  init_kernel<<<(B_ * S_ + 255) / 256, 256, 0, stream>>>(mask, dec_x);

  size_t encSmem = (size_t)G4 * WP * 2 + (size_t)H_ * WP * 2 +
                   (size_t)2 * 16 * WP * 2 + (size_t)XC * 16 * 4;
  enc_kernel<<<16, 256, encSmem, stream>>>(x, eWih, eWhh, ebih, ebhh, W1w, W1b,
                                           keys, hx, cx);

  size_t decSmem = (size_t)G4 * WP * 2 + (size_t)H_ * WP * 2 + (size_t)16 * WP * 2;
  size_t attnSmem = (size_t)256 * WP * 2 + (size_t)(H_ + H_ + 256) * 4 +
                    (size_t)256 * 4;
  for (int step = 0; step < T_; ++step) {
    dec_step_kernel<<<16, 256, decSmem, stream>>>(dWih, dWhh, dbih, dbhh, W2w,
                                                  W2b, dec_x, hx, cx, query);
    attn_kernel<<<B_, 256, attnSmem, stream>>>(x, keys, query, vtw, vtb, mask,
                                               dec_x, out, step);
  }
}
